// LinearLoopLayer_19284403159518
// MI455X (gfx1250) — matrix operand order repaired and checked by reading, not compiled, not run on hardware
//
#include <hip/hip_runtime.h>

// MI455X / gfx1250: wave32, WMMA 16x16x32 bf16 with f32 accumulate.
// f32 GEMM emulated via bf16 hi/lo split (3 WMMA products: hh + hl + lh),
// giving ~fp32 accuracy at ~3/8 the matrix-instruction count of the
// V_WMMA_F32_16X16X4_F32 path.

typedef __attribute__((ext_vector_type(16))) __bf16 v16bf;
typedef __attribute__((ext_vector_type(8)))  __bf16 v8bf;
typedef __attribute__((ext_vector_type(4)))  __bf16 v4bf;
typedef __attribute__((ext_vector_type(8)))  float  v8f;

#define KC 32            // K-chunk per LDS stage == WMMA K
#define LDT 40           // row stride in bf16 (32 + 8 pad -> conflict-free b128)

// Load a [ROWS x 32] f32 tile (row-major, leading dim srcLd) from global,
// split into bf16 hi/lo, store to padded LDS tiles.
template<int ROWS, int NTHR>
__device__ inline void load_split_tile(const float* __restrict__ src, int srcLd,
                                       __bf16* shi, __bf16* slo, int tid, bool pf) {
    constexpr int NV = ROWS * KC / 4;          // float4 elements
    static_assert(NV % NTHR == 0, "tile/threads mismatch");
#pragma unroll
    for (int it = 0; it < NV / NTHR; ++it) {
        int i = it * NTHR + tid;
        int r = i >> 3;                        // / (KC/4)
        int c = (i & 7) << 2;                  // * 4
        const float* p = src + (size_t)r * srcLd + c;
        float4 v = *(const float4*)p;
        if (pf) __builtin_prefetch(p + KC, 0, 3);   // next K-chunk -> global_prefetch_b8
        v4bf h, l;
        h[0] = (__bf16)v.x; l[0] = (__bf16)(v.x - (float)h[0]);
        h[1] = (__bf16)v.y; l[1] = (__bf16)(v.y - (float)h[1]);
        h[2] = (__bf16)v.z; l[2] = (__bf16)(v.z - (float)h[2]);
        h[3] = (__bf16)v.w; l[3] = (__bf16)(v.w - (float)h[3]);
        int o = r * LDT + c;
        *(v4bf*)(shi + o) = h;
        *(v4bf*)(slo + o) = l;
    }
}

// A fragment, 16-bit A-matrix 16x32 layout: lane L holds row M=L%16;
// vector elems 0-7  = K =      8*(L/16) + 0..7
// vector elems 8-15 = K = 16 + 8*(L/16) + 0..7
__device__ inline v16bf load_fragA(const __bf16* s, int row, int half) {
    const __bf16* p = s + row * LDT;
    v8bf a = *(const v8bf*)(p + 8 * half);
    v8bf b = *(const v8bf*)(p + 16 + 8 * half);
    return __builtin_shufflevector(a, b, 0,1,2,3,4,5,6,7,8,9,10,11,12,13,14,15);
}

// B fragment, 16-bit B-matrix 32x16 layout: lane L holds col N=L%16;
// vector elems 0-7  = K =      8*(L/16) + 0..7
// vector elems 8-15 = K = 16 + 8*(L/16) + 0..7
// Our B source is Bm^T with Bm row-major [N x K], so this is a contiguous row read.
__device__ inline v16bf load_fragB(const __bf16* s, int col, int half) {
    const __bf16* p = s + col * LDT + 8 * half;
    v8bf a = *(const v8bf*)(p);
    v8bf b = *(const v8bf*)(p + 16);
    return __builtin_shufflevector(a, b, 0,1,2,3,4,5,6,7,8,9,10,11,12,13,14,15);
}

__device__ inline v8f wmma_bf16(v16bf a, v16bf b, v8f c) {
    return __builtin_amdgcn_wmma_f32_16x16x32_bf16(
        /*neg_a=*/false, a, /*neg_b=*/false, b,
        /*c_mod=*/(short)0, c, /*reuse_a=*/false, /*reuse_b=*/false);
}

// D[M x N] = A[M x K] * Bm[N x K]^T (+ bias), all f32 row-major. M,N,K % tile == 0.
template<int BM, int BN, int WMW, int WNW, int TM, int TN, bool BIAS>
__global__ __launch_bounds__(WMW * WNW * 32)
void gemm_lowrank_bf16x3(const float* __restrict__ A, const float* __restrict__ Bm,
                         const float* __restrict__ bias, float* __restrict__ D,
                         int N, int K) {
    static_assert(WMW * TM * 16 == BM && WNW * TN * 16 == BN, "tiling mismatch");
    constexpr int NTHR = WMW * WNW * 32;

    __shared__ __bf16 sAhi[BM * LDT], sAlo[BM * LDT];
    __shared__ __bf16 sBhi[BN * LDT], sBlo[BN * LDT];

    const int tid  = threadIdx.x;
    const int lane = tid & 31;          // wave32
    const int wave = tid >> 5;
    const int wm   = (wave / WNW) * (TM * 16);
    const int wn   = (wave % WNW) * (TN * 16);
    const int half = lane >> 4;
    const int l16  = lane & 15;

    const int row0 = blockIdx.y * BM;
    const int col0 = blockIdx.x * BN;

    v8f acc[TM][TN];
#pragma unroll
    for (int i = 0; i < TM; ++i)
#pragma unroll
        for (int j = 0; j < TN; ++j) acc[i][j] = (v8f){};

    for (int k0 = 0; k0 < K; k0 += KC) {
        const bool pf = (k0 + KC) < K;
        load_split_tile<BM, NTHR>(A  + (size_t)row0 * K + k0, K, sAhi, sAlo, tid, pf);
        load_split_tile<BN, NTHR>(Bm + (size_t)col0 * K + k0, K, sBhi, sBlo, tid, pf);
        __syncthreads();

        v16bf ahi[TM], alo[TM], bhi[TN], blo[TN];
#pragma unroll
        for (int i = 0; i < TM; ++i) {
            ahi[i] = load_fragA(sAhi, wm + i * 16 + l16, half);
            alo[i] = load_fragA(sAlo, wm + i * 16 + l16, half);
        }
#pragma unroll
        for (int j = 0; j < TN; ++j) {
            bhi[j] = load_fragB(sBhi, wn + j * 16 + l16, half);
            blo[j] = load_fragB(sBlo, wn + j * 16 + l16, half);
        }
#pragma unroll
        for (int i = 0; i < TM; ++i)
#pragma unroll
            for (int j = 0; j < TN; ++j) {
                acc[i][j] = wmma_bf16(ahi[i], bhi[j], acc[i][j]);  // hi*hi
                acc[i][j] = wmma_bf16(ahi[i], blo[j], acc[i][j]);  // hi*lo
                acc[i][j] = wmma_bf16(alo[i], bhi[j], acc[i][j]);  // lo*hi
            }
        __syncthreads();
    }

    // C/D f32 16x16 layout: VGPR v -> M = v + 8*half, N = lane%16
#pragma unroll
    for (int i = 0; i < TM; ++i)
#pragma unroll
        for (int j = 0; j < TN; ++j) {
            const int gc = col0 + wn + j * 16 + l16;
            float b = BIAS ? bias[gc] : 0.0f;
#pragma unroll
            for (int v = 0; v < 8; ++v) {
                const int gr = row0 + wm + i * 16 + v + 8 * half;
                D[(size_t)gr * N + gc] = acc[i][j][v] + b;
            }
        }
}

extern "C" void kernel_launch(void* const* d_in, const int* in_sizes, int n_in,
                              void* d_out, int out_size, void* d_ws, size_t ws_size,
                              hipStream_t stream) {
    const float* x    = (const float*)d_in[0];   // [4096, 4096]
    const float* U    = (const float*)d_in[1];   // [256, 4096]
    const float* V    = (const float*)d_in[2];   // [4096, 256]
    const float* bias = (const float*)d_in[3];   // [4096]
    float* out = (float*)d_out;                  // [4096, 4096]
    float* T   = (float*)d_ws;                   // [4096, 256] f32 scratch (4 MB)

    constexpr int B = 4096, IN = 4096, OUTD = 4096, R = 256;

    // GEMM1: T = x * U^T  (M=4096, N=256, K=4096). 64x64 blocks -> 256 WGs.
    {
        dim3 grid(R / 64, B / 64);
        gemm_lowrank_bf16x3<64, 64, 2, 4, 2, 1, false>
            <<<grid, 256, 0, stream>>>(x, U, nullptr, T, R, IN);
    }
    // GEMM2: out = T * V^T + bias  (M=4096, N=4096, K=256). 128x128 blocks -> 1024 WGs.
    {
        dim3 grid(OUTD / 128, B / 128);
        gemm_lowrank_bf16x3<128, 128, 2, 4, 4, 2, true>
            <<<grid, 256, 0, stream>>>(T, V, bias, out, OUTD, R);
    }
}
